// GatedGraphConvModule_38886633898060
// MI455X (gfx1250) — matrix operand order repaired and checked by reading, not compiled, not run on hardware
//
#include <hip/hip_runtime.h>
#include <math.h>

#define N_NODES  50000
#define N_EDGES  800000
#define DIM      64
#define N_ETYPES 4
#define N_STEPS  5
#define ROW_TILES (N_NODES / 16)   // 3125 (N_NODES divisible by 16)

typedef __attribute__((ext_vector_type(16))) _Float16 v16h;
typedef __attribute__((ext_vector_type(8)))  float    v8f;

// ---------------------------------------------------------------------------
// WMMA fragment layouts (wave32, v_wmma_f32_16x16x32_f16)
//
// A (16x32, MxK): lane = half*16 + m (m=row),
//   half-word i in 0..15 holds K = half*16 + i -> contiguous in K (same as B)
// B (32x16, KxN): lane = half*16 + n (n=col); half-word i in 0..15 holds
//   K = half*16 + i -> contiguous in K.
// C/D (16x16 f32): VGPR r holds row = half*8 + r, col = lane&15.
// ---------------------------------------------------------------------------

__device__ __forceinline__ v16h load_a_frag(const float* __restrict__ X,
                                            int row, int ld, int kb, int half) {
  const float* p = X + row * ld + kb;
  v16h a;
#pragma unroll
  for (int i = 0; i < 16; ++i) {
    const int v  = i >> 1;
    const int hi = i & 1;
    const int k  = half * 16 + i;
    a[i] = (_Float16)p[k];
  }
  return a;
}

// Pre-packed B fragment: frag layout is [fragIdx][lane][16 halfs] -> one
// contiguous 32B load per lane, fully coalesced across the wave.
__device__ __forceinline__ v16h load_b_frag_packed(const _Float16* __restrict__ P,
                                                   int fragIdx, int lane) {
  return *(const v16h*)(P + (((fragIdx << 5) + lane) << 4));
}

__device__ __forceinline__ v8f wmma_acc(v16h a, v16h b, v8f c) {
  return __builtin_amdgcn_wmma_f32_16x16x32_f16(
      /*neg_a=*/false, a, /*neg_b=*/false, b,
      /*c_mod=*/(short)0, c, /*reuse_a=*/false, /*reuse_b=*/false);
}

__device__ __forceinline__ float sigmoid_f(float x) {
  return 1.0f / (1.0f + __expf(-x));
}

// ---------------------------------------------------------------------------
// Weight pre-pack: fp32 row-major [R x 64] -> f16 B-fragments.
// Frag order: fragIdx = tile*2 + kbI  (tile = out-col tile of 16, kbI = K/32).
// Element (fragIdx, lane, i) <- src[(tile*16 + n)*64 + kb + half*16 + i],
// n = lane&15, half = lane>>4. Bijection over all R*64 elements.
// ---------------------------------------------------------------------------
__global__ void pack_weights_kernel(const float* __restrict__ src,
                                    _Float16* __restrict__ dst, int total) {
  const int gid = blockIdx.x * blockDim.x + threadIdx.x;
  if (gid >= total) return;
  const int fragIdx = gid >> 9;            // 512 halfs per fragment
  const int rem     = gid & 511;
  const int lane    = rem >> 4;
  const int i       = rem & 15;
  const int tile    = fragIdx >> 1;
  const int kb      = (fragIdx & 1) << 5;
  const int n       = lane & 15;
  const int half    = lane >> 4;
  dst[gid] = (_Float16)src[(tile * 16 + n) * DIM + kb + half * 16 + i];
}

// ---------------------------------------------------------------------------
// Kernel 1: h_all[n, t, :] = W[t] @ h[n] + b[t]   (4 GEMMs [N x 64]·[64 x 64])
// One wave = one 16-row tile of one etype. 8 waves/block = 2 tiles x 4 etypes.
// ---------------------------------------------------------------------------
__global__ void __launch_bounds__(256)
etype_linear_kernel(const float* __restrict__ h, const _Float16* __restrict__ pW,
                    const float* __restrict__ bias, float* __restrict__ h_all) {
  const int wave = threadIdx.x >> 5;
  const int lane = threadIdx.x & 31;
  const int t    = wave & 3;
  const int tile = blockIdx.x * 2 + (wave >> 2);
  if (tile >= ROW_TILES) return;            // uniform per wave -> EXEC all-1s

  const int row0 = tile * 16;
  const int m    = lane & 15;               // row (A) / col (B,C)
  const int half = lane >> 4;

  const v16h a0 = load_a_frag(h, row0 + m, DIM, 0,  half);
  const v16h a1 = load_a_frag(h, row0 + m, DIM, 32, half);

#pragma unroll
  for (int c = 0; c < 4; ++c) {
    const int col0 = c * 16;
    const float bv = bias[t * DIM + col0 + m];
    v8f acc;
#pragma unroll
    for (int r = 0; r < 8; ++r) acc[r] = bv;
    const int fbase = (t * 4 + c) << 1;     // W packed as R=256 rows (t*64+e)
    acc = wmma_acc(a0, load_b_frag_packed(pW, fbase + 0, lane), acc);
    acc = wmma_acc(a1, load_b_frag_packed(pW, fbase + 1, lane), acc);
#pragma unroll
    for (int r = 0; r < 8; ++r) {
      const int row = row0 + half * 8 + r;
      h_all[row * (N_ETYPES * DIM) + t * DIM + col0 + m] = acc[r];
    }
  }
}

// ---------------------------------------------------------------------------
// Kernel 2: zero / copy helpers (float4-wide)
// ---------------------------------------------------------------------------
__global__ void zero4_kernel(float4* __restrict__ p, int n4) {
  const int i = blockIdx.x * blockDim.x + threadIdx.x;
  if (i < n4) p[i] = make_float4(0.f, 0.f, 0.f, 0.f);
}

__global__ void copy4_kernel(const float4* __restrict__ s, float4* __restrict__ d, int n4) {
  const int i = blockIdx.x * blockDim.x + threadIdx.x;
  if (i < n4) d[i] = s[i];
}

// ---------------------------------------------------------------------------
// Kernel 3: edge gather + scatter-add. 64 threads per edge (one per dim):
// coalesced gather of h_all[src, etype, :], f32 atomics into L2-resident acc.
// ---------------------------------------------------------------------------
__global__ void __launch_bounds__(256)
edge_scatter_kernel(const float* __restrict__ h_all, const int* __restrict__ src,
                    const int* __restrict__ dst, const int* __restrict__ ety,
                    float* __restrict__ acc) {
  const int gid = blockIdx.x * blockDim.x + threadIdx.x;
  if (gid >= N_EDGES * DIM) return;
  const int e = gid >> 6;
  const int d = gid & 63;
  const int s = src[e];
  const int t = ety[e];
  const float v = h_all[s * (N_ETYPES * DIM) + t * DIM + d];
  atomicAdd(&acc[dst[e] * DIM + d], v);
}

// ---------------------------------------------------------------------------
// Kernel 4: GRU cell. gi = a @ w_ih^T + b_ih, gh = h @ w_hh^T + b_hh (K=64,
// 192 outputs each) with WMMA; gate math elementwise on aligned fragments;
// h updated in place (each wave owns its 16 rows exclusively).
// ---------------------------------------------------------------------------
__global__ void __launch_bounds__(256)
gru_kernel(const float* __restrict__ a, float* __restrict__ h,
           const _Float16* __restrict__ pWih, const _Float16* __restrict__ pWhh,
           const float* __restrict__ b_ih, const float* __restrict__ b_hh) {
  const int wave = threadIdx.x >> 5;
  const int lane = threadIdx.x & 31;
  const int tile = blockIdx.x * 8 + wave;
  if (tile >= ROW_TILES) return;

  const int row0 = tile * 16;
  const int m    = lane & 15;
  const int half = lane >> 4;

  const v16h Aa0 = load_a_frag(a, row0 + m, DIM, 0,  half);
  const v16h Aa1 = load_a_frag(a, row0 + m, DIM, 32, half);
  const v16h Ah0 = load_a_frag(h, row0 + m, DIM, 0,  half);
  const v16h Ah1 = load_a_frag(h, row0 + m, DIM, 32, half);

#pragma unroll
  for (int c = 0; c < 4; ++c) {
    const int col0 = c * 16;
    v8f gi[3], gh[3];
#pragma unroll
    for (int g = 0; g < 3; ++g) {            // gates r, z, n
      const int ocol  = g * DIM + col0;      // column block in 192-wide output
      const int fbase = (g * 4 + c) << 1;    // fragment index for this col tile
      const float bi = b_ih[ocol + m];
      const float bh = b_hh[ocol + m];
      v8f ci, ch;
#pragma unroll
      for (int r = 0; r < 8; ++r) { ci[r] = bi; ch[r] = bh; }
      ci = wmma_acc(Aa0, load_b_frag_packed(pWih, fbase + 0, lane), ci);
      ci = wmma_acc(Aa1, load_b_frag_packed(pWih, fbase + 1, lane), ci);
      ch = wmma_acc(Ah0, load_b_frag_packed(pWhh, fbase + 0, lane), ch);
      ch = wmma_acc(Ah1, load_b_frag_packed(pWhh, fbase + 1, lane), ch);
      gi[g] = ci; gh[g] = ch;
    }
#pragma unroll
    for (int r = 0; r < 8; ++r) {
      const int idx  = (row0 + half * 8 + r) * DIM + col0 + m;
      const float ho = h[idx];
      const float rg = sigmoid_f(gi[0][r] + gh[0][r]);
      const float zg = sigmoid_f(gi[1][r] + gh[1][r]);
      const float ng = tanhf(gi[2][r] + rg * gh[2][r]);
      h[idx] = (1.0f - zg) * ng + zg * ho;
    }
  }
}

// ---------------------------------------------------------------------------
// Host launcher
// ---------------------------------------------------------------------------
extern "C" void kernel_launch(void* const* d_in, const int* in_sizes, int n_in,
                              void* d_out, int out_size, void* d_ws, size_t ws_size,
                              hipStream_t stream) {
  const float* feat = (const float*)d_in[0];
  const float* W    = (const float*)d_in[1];
  const float* bias = (const float*)d_in[2];
  const float* w_ih = (const float*)d_in[3];
  const float* w_hh = (const float*)d_in[4];
  const float* b_ih = (const float*)d_in[5];
  const float* b_hh = (const float*)d_in[6];
  const int*   src  = (const int*)d_in[7];
  const int*   dst  = (const int*)d_in[8];
  const int*   ety  = (const int*)d_in[9];

  float* h     = (float*)d_out;                          // [N, DIM], in place
  float* a_buf = (float*)d_ws;                           // [N, DIM] accumulator
  float* h_all = a_buf + (size_t)N_NODES * DIM;          // [N, T, DIM]
  _Float16* pW   = (_Float16*)(h_all + (size_t)N_NODES * N_ETYPES * DIM);
  _Float16* pWih = pW   + 256 * DIM;                     // W packed: 256 rows
  _Float16* pWhh = pWih + 192 * DIM;                     // w_ih packed: 192 rows

  const int n4_h = N_NODES * DIM / 4;                    // 800000 float4
  const int nsct = N_EDGES * DIM;                        // 51.2M threads

  // One-shot: pack weights into f16 B-fragment layout (loop invariant).
  pack_weights_kernel<<<(256 * DIM + 255) / 256, 256, 0, stream>>>(W,    pW,   256 * DIM);
  pack_weights_kernel<<<(192 * DIM + 255) / 256, 256, 0, stream>>>(w_ih, pWih, 192 * DIM);
  pack_weights_kernel<<<(192 * DIM + 255) / 256, 256, 0, stream>>>(w_hh, pWhh, 192 * DIM);

  // h = feat
  copy4_kernel<<<(n4_h + 255) / 256, 256, 0, stream>>>((const float4*)feat,
                                                       (float4*)h, n4_h);

  for (int step = 0; step < N_STEPS; ++step) {
    etype_linear_kernel<<<(ROW_TILES + 1) / 2, 256, 0, stream>>>(h, pW, bias, h_all);
    zero4_kernel<<<(n4_h + 255) / 256, 256, 0, stream>>>((float4*)a_buf, n4_h);
    edge_scatter_kernel<<<(nsct + 255) / 256, 256, 0, stream>>>(h_all, src, dst, ety, a_buf);
    gru_kernel<<<(ROW_TILES + 7) / 8, 256, 0, stream>>>(a_buf, h, pWih, pWhh, b_ih, b_hh);
  }
}
